// ContTimeSpaceLSTMCell_39719857553628
// MI455X (gfx1250) — compile-verified, matrix operand order checked by reading, not run on hardware
//
#include <hip/hip_runtime.h>

// ---------------------------------------------------------------------------
// ContTimeSpaceLSTMCell fused kernel for MI455X (gfx1250, wave32, WMMA)
// B=8192, H=1024, NC=3, K=2H=2048, N_total=10H=10240
// Round 4: pin ds_load(t+1) above wmma(t) with sched_barrier so the wait
//          inserter can use partial dscnt waits (true fragment pipelining).
// ---------------------------------------------------------------------------

typedef __attribute__((ext_vector_type(16))) __bf16 v16bf;
typedef __attribute__((ext_vector_type(8)))  float  v8f;

#define B_DIM 8192
#define H_DIM 1024
#define K_DIM 2048
#define BM    128
#define KC    64
#define A_LD  72   // padded LDS row stride (elements) to avoid bank conflicts
#define B_LD  72

// ---- f32 -> bf16 conversion / packing kernels -----------------------------

__global__ void cvt_xh_kernel(const float* __restrict__ x,
                              const float* __restrict__ h,
                              __bf16* __restrict__ xh) {
    int idx = blockIdx.x * blockDim.x + threadIdx.x;      // 0 .. B*2H-1
    int b = idx >> 11;
    int c = idx & 2047;
    float v = (c < H_DIM) ? x[b * H_DIM + c] : h[b * H_DIM + (c - H_DIM)];
    xh[idx] = (__bf16)v;
}

__global__ void cvt_w_kernel(const float* __restrict__ w0, const float* __restrict__ w1,
                             const float* __restrict__ w2, const float* __restrict__ w3,
                             const float* __restrict__ w4, const float* __restrict__ w5,
                             const float* __restrict__ wd, __bf16* __restrict__ Wc) {
    int idx = blockIdx.x * blockDim.x + threadIdx.x;      // 0 .. 10H*2H-1
    int r = idx >> 11;
    int c = idx & 2047;
    int g = r >> 10;
    int rr = r & 1023;
    float v;
    switch (g) {
        case 0: v = w0[rr * K_DIM + c]; break;
        case 1: v = w1[rr * K_DIM + c]; break;
        case 2: v = w2[rr * K_DIM + c]; break;
        case 3: v = w3[rr * K_DIM + c]; break;
        case 4: v = w4[rr * K_DIM + c]; break;
        case 5: v = w5[rr * K_DIM + c]; break;
        default: v = wd[(r - 6144) * K_DIM + c]; break;
    }
    Wc[idx] = (__bf16)v;
}

__global__ void cvt_b_kernel(const float* __restrict__ b0, const float* __restrict__ b1,
                             const float* __restrict__ b2, const float* __restrict__ b3,
                             const float* __restrict__ b4, const float* __restrict__ b5,
                             const float* __restrict__ bd, float* __restrict__ bc) {
    int r = blockIdx.x * blockDim.x + threadIdx.x;        // 0 .. 10239
    if (r >= 10240) return;
    int g = r >> 10;
    int rr = r & 1023;
    float v;
    switch (g) {
        case 0: v = b0[rr]; break;
        case 1: v = b1[rr]; break;
        case 2: v = b2[rr]; break;
        case 3: v = b3[rr]; break;
        case 4: v = b4[rr]; break;
        case 5: v = b5[rr]; break;
        default: v = bd[r - 6144]; break;
    }
    bc[r] = v;
}

// ---- device helpers -------------------------------------------------------

__device__ __forceinline__ void sched_fence() {
#if __has_builtin(__builtin_amdgcn_sched_barrier)
    __builtin_amdgcn_sched_barrier(0);    // no instruction may cross
#endif
}

__device__ __forceinline__ float fast_tanh(float x) {
#if __has_builtin(__builtin_amdgcn_tanhf)
    return __builtin_amdgcn_tanhf(x);     // v_tanh_f32 (CDNA5 TRANS op)
#else
    return tanhf(x);
#endif
}
__device__ __forceinline__ float sig_f(float x) {
    // sigmoid(x) = 0.5 * tanh(x/2) + 0.5  -> one TRANS + FMA
    return __builtin_fmaf(0.5f, fast_tanh(0.5f * x), 0.5f);
}
__device__ __forceinline__ float splus_f(float x) {
    // softplus, BETA = 1
    return (x > 15.0f) ? x : log1pf(__expf(x));
}

// CDNA5 async global->LDS copy: 16 bytes per lane, tracked by ASYNCcnt.
// lds_off = LDS byte offset (low 32 bits of a generic pointer to __shared__,
// per ISA 10.2); gaddr = 64-bit global address.
__device__ __forceinline__ void async_b128(unsigned lds_off, const void* gaddr) {
    asm volatile("global_load_async_to_lds_b128 %0, %1, off"
                 :: "v"(lds_off), "v"((unsigned long long)gaddr)
                 : "memory");
}
__device__ __forceinline__ void wait_async0() {
    asm volatile("s_wait_asynccnt 0x0" ::: "memory");
}

// ---- fused GEMM + gate epilogue -------------------------------------------
// grid: (B/BM = 64, H/16 = 64), block: 256 (8 waves)
// Wave w computes a 16x160 strip: 6 gate tiles (cols h0+g*H .. +15) and
// 4 decay tiles (cols 6H + 4*h0 .. +63), all over K=2048.

__device__ __forceinline__ void stage_async(const __bf16* __restrict__ xh,
                                            const __bf16* __restrict__ Wc,
                                            int tid, int rbase, int h0, int kb,
                                            unsigned aBase, unsigned bBase) {
    // A tile: 128 x 64 bf16 = 1024 x 16B, 4 per thread
    #pragma unroll
    for (int i = 0; i < 4; ++i) {
        int ii = tid + i * 256;
        int ar = ii >> 3;
        int ac = ii & 7;
        const void* g = (const void*)(xh + (size_t)(rbase + ar) * K_DIM + kb + ac * 8);
        async_b128(aBase + (unsigned)(ar * A_LD + ac * 8) * 2u, g);
    }
    // B tile: 160 x 64 bf16 = 1280 x 16B, 5 per thread
    #pragma unroll
    for (int i = 0; i < 5; ++i) {
        int ii = tid + i * 256;
        int br = ii >> 3;
        int bcc = ii & 7;
        int grow = (br < 96) ? (h0 + (br >> 4) * H_DIM + (br & 15))
                             : (6144 + 4 * h0 + (br - 96));
        const void* g = (const void*)(Wc + (size_t)grow * K_DIM + kb + bcc * 8);
        async_b128(bBase + (unsigned)(br * B_LD + bcc * 8) * 2u, g);
    }
}

__launch_bounds__(256, 2)
__global__ void lstm_wmma_kernel(const __bf16* __restrict__ xh,
                                 const __bf16* __restrict__ Wc,
                                 const float*  __restrict__ bc,
                                 const float*  __restrict__ cprev,
                                 const float*  __restrict__ cbprev,
                                 float* __restrict__ out) {
    __shared__ __align__(16) __bf16 As[2][BM * A_LD];    // 2 x 128 x 64 (padded)
    __shared__ __align__(16) __bf16 Bs[2][160 * B_LD];   // 2 x 160 x 64 (padded)

    const int tid   = threadIdx.x;
    const int w     = tid >> 5;
    const int lane  = tid & 31;
    const int hi    = lane >> 4;     // half-wave select
    const int l15   = lane & 15;
    const int rbase = blockIdx.x * BM;
    const int h0    = blockIdx.y * 16;

    const unsigned asb0 = (unsigned)(size_t)(void*)&As[0][0];
    const unsigned asb1 = (unsigned)(size_t)(void*)&As[1][0];
    const unsigned bsb0 = (unsigned)(size_t)(void*)&Bs[0][0];
    const unsigned bsb1 = (unsigned)(size_t)(void*)&Bs[1][0];

    v8f acc[10] = {};

    // Prologue: fill buffer 0
    stage_async(xh, Wc, tid, rbase, h0, 0, asb0, bsb0);

    for (int kb = 0; kb < K_DIM; kb += KC) {
        const int buf = (kb >> 6) & 1;

        wait_async0();        // my async writes for current buffer are in LDS
        __syncthreads();      // everyone's are; prev compute on other buf done

        if (kb + KC < K_DIM) {
            stage_async(xh, Wc, tid, rbase, h0, kb + KC,
                        buf ? asb0 : asb1, buf ? bsb0 : bsb1);
        }

        const __bf16* Abuf = &As[buf][0];
        const __bf16* Bbuf = &Bs[buf][0];

        // Two k-steps of 32 per staged chunk
        #pragma unroll
        for (int s = 0; s < 2; ++s) {
            union U { uint4 u[2]; v16bf v; };
            U af, bf0, bf1;
            // A fragment (16-bit 16x32 layout: lanes0-15 K0-7/16-23, lanes16-31 K8-15/24-31)
            const __bf16* ap = Abuf + (w * 16 + l15) * A_LD + s * 32 + hi * 8;
            af.u[0] = *(const uint4*)ap;
            af.u[1] = *(const uint4*)(ap + 16);
            // Preload B fragment for tile 0
            {
                const __bf16* bp = Bbuf + l15 * B_LD + s * 32 + hi * 8;
                bf0.u[0] = *(const uint4*)bp;
                bf0.u[1] = *(const uint4*)(bp + 16);
            }
            // Pipelined sweep over the 10 N-tiles: loads for tile t+1 are
            // pinned ABOVE the WMMA of tile t by a scheduling barrier, so the
            // wait before WMMA t+1 is a partial dscnt wait and the LDS latency
            // hides behind one WMMA.
            #pragma unroll
            for (int t = 0; t < 10; ++t) {
                U& cur = (t & 1) ? bf1 : bf0;
                U& nxt = (t & 1) ? bf0 : bf1;
                if (t < 9) {
                    const __bf16* bp = Bbuf + ((t + 1) * 16 + l15) * B_LD + s * 32 + hi * 8;
                    nxt.u[0] = *(const uint4*)bp;
                    nxt.u[1] = *(const uint4*)(bp + 16);
                }
                sched_fence();   // keep the nxt loads above this WMMA
                acc[t] = __builtin_amdgcn_wmma_f32_16x16x32_bf16(
                    /*neg_a=*/false, af.v, /*neg_b=*/false, cur.v,
                    /*c_mod=*/(short)0, acc[t],
                    /*reuse_a=*/false, /*reuse_b=*/false);
            }
        }
    }

    // ---- epilogue: gates + cell updates + softplus decay ----
    const int   c   = l15;
    const int   hc  = h0 + c;
    const float bi  = bc[hc];
    const float bfg = bc[1024 + hc];
    const float bo  = bc[2048 + hc];
    const float bib = bc[3072 + hc];
    const float bfb = bc[4096 + hc];
    const float bpc = bc[5120 + hc];
    const float bd0 = bc[6144 + 4 * h0 + c];
    const float bd1 = bc[6144 + 4 * h0 + 16 + c];
    const float bd2 = bc[6144 + 4 * h0 + 32 + c];
    const float bd3 = bc[6144 + 4 * h0 + 48 + c];

    const int    r0 = rbase + w * 16 + hi * 8;
    const size_t BH = (size_t)B_DIM * H_DIM;

    #pragma unroll
    for (int j = 0; j < 8; ++j) {
        int    r  = r0 + j;                       // C/D layout: VGPR j -> M = j + 8*hi
        size_t ro = (size_t)r * H_DIM + hc;

        float gi  = sig_f(acc[0][j] + bi);
        float gf  = sig_f(acc[1][j] + bfg);
        float go  = sig_f(acc[2][j] + bo);
        float gib = sig_f(acc[3][j] + bib);
        float gfb = sig_f(acc[4][j] + bfb);
        float pc  = fast_tanh(acc[5][j] + bpc);

        float cp  = cprev[ro];
        float cbp = cbprev[ro];

        out[ro]            = gf  * cp  + gi  * pc;   // cell_i
        out[BH + ro]       = gfb * cbp + gib * pc;   // cell_bar_i
        out[6 * BH + ro]   = go;                     // go

        size_t dbase = 2 * BH + (size_t)r * 4096 + 4 * h0 + c;   // gate_decay
        out[dbase]      = splus_f(acc[6][j] + bd0);
        out[dbase + 16] = splus_f(acc[7][j] + bd1);
        out[dbase + 32] = splus_f(acc[8][j] + bd2);
        out[dbase + 48] = splus_f(acc[9][j] + bd3);
    }
}

// ---------------------------------------------------------------------------

extern "C" void kernel_launch(void* const* d_in, const int* in_sizes, int n_in,
                              void* d_out, int out_size, void* d_ws, size_t ws_size,
                              hipStream_t stream) {
    const float* x_i      = (const float*)d_in[0];
    const float* hidden   = (const float*)d_in[1];
    const float* cell_m   = (const float*)d_in[2];
    const float* cellb_m  = (const float*)d_in[3];
    const float* W_input  = (const float*)d_in[4];
    const float* b_input  = (const float*)d_in[5];
    const float* W_forget = (const float*)d_in[6];
    const float* b_forget = (const float*)d_in[7];
    const float* W_output = (const float*)d_in[8];
    const float* b_output = (const float*)d_in[9];
    const float* W_ibar   = (const float*)d_in[10];
    const float* b_ibar   = (const float*)d_in[11];
    const float* W_fbar   = (const float*)d_in[12];
    const float* b_fbar   = (const float*)d_in[13];
    const float* W_prec   = (const float*)d_in[14];
    const float* b_prec   = (const float*)d_in[15];
    const float* W_decay  = (const float*)d_in[16];
    const float* b_decay  = (const float*)d_in[17];

    // workspace layout: xh_bf16 [B,2H] | Wcat_bf16 [10H,2H] | bcat_f32 [10H]
    const size_t XH_BYTES = (size_t)B_DIM * K_DIM * sizeof(__bf16);   // 33.5 MB
    const size_t W_BYTES  = (size_t)10240 * K_DIM * sizeof(__bf16);   // 41.9 MB
    __bf16* xh = (__bf16*)d_ws;
    __bf16* Wc = (__bf16*)((char*)d_ws + XH_BYTES);
    float*  bc = (float*)((char*)d_ws + XH_BYTES + W_BYTES);

    cvt_xh_kernel<<<(B_DIM * K_DIM) / 256, 256, 0, stream>>>(x_i, hidden, xh);
    cvt_w_kernel<<<(10240 * K_DIM) / 256, 256, 0, stream>>>(
        W_input, W_forget, W_output, W_ibar, W_fbar, W_prec, W_decay, Wc);
    cvt_b_kernel<<<40, 256, 0, stream>>>(
        b_input, b_forget, b_output, b_ibar, b_fbar, b_prec, b_decay, bc);

    dim3 grid(B_DIM / BM, H_DIM / 16);
    lstm_wmma_kernel<<<grid, 256, 0, stream>>>(xh, Wc, bc, cell_m, cellb_m,
                                               (float*)d_out);
}
